// Decoder_21543555957003
// MI455X (gfx1250) — matrix operand order repaired and checked by reading, not compiled, not run on hardware
//
#include <hip/hip_runtime.h>
#include <hip/hip_bf16.h>

// ---------------------------------------------------------------------------
// Types for CDNA5 WMMA / TDM (gfx1250, wave32)
// ---------------------------------------------------------------------------
typedef __attribute__((ext_vector_type(16))) __bf16 v16bf;
typedef __attribute__((ext_vector_type(8)))  __bf16 v8bf;
typedef __attribute__((ext_vector_type(8)))  float  v8f;
typedef __attribute__((ext_vector_type(4)))  float  v4f;
typedef __attribute__((ext_vector_type(4)))  unsigned int tdm_u32x4;
typedef __attribute__((ext_vector_type(8)))  int          tdm_i32x8;
typedef __attribute__((ext_vector_type(4)))  int          tdm_i32x4;

static __device__ __forceinline__ float sigmoidf_(float x) {
    return 1.0f / (1.0f + __expf(-x));
}

// ---------------------------------------------------------------------------
// TDM: issue a 2D tile load (64 cols x 64 rows of bf16) from global into LDS.
// D# layout per CDNA5 ISA 8.3/8.4:
//   group0: [1:0]=count=1, [63:32]=lds_addr, [120:64]=global_addr, [127:126]=2
//   group1: data_size=1(2B), tensor_dim0/1 (OOB), tile_dim0=64, tile_dim1=64,
//           tensor_dim0_stride=N
//   group2/3 (+ trailing group): zero (2D tensor)
// ---------------------------------------------------------------------------
static __device__ __forceinline__ void tdm_load_slab(
    const __bf16* gbase, unsigned ldsOff, int width, int rows, int N)
{
    unsigned long long ga = (unsigned long long)(uintptr_t)gbase;
    tdm_u32x4 g0;
    g0[0] = 1u;                                            // count = 1
    g0[1] = ldsOff;                                        // lds_addr (bytes)
    g0[2] = (unsigned)ga;                                  // global_addr lo
    g0[3] = ((unsigned)(ga >> 32) & 0x01FFFFFFu) | 0x80000000u;  // hi | type=2
    tdm_i32x8 g1;
    g1[0] = 0x00010000;                                    // data_size = 2 bytes
    g1[1] = (int)((unsigned)(width & 0xFFFF) << 16);       // tensor_dim0 lo16
    g1[2] = (int)(((unsigned)(width >> 16) & 0xFFFFu) |
                  ((unsigned)(rows & 0xFFFF) << 16));      // dim0 hi | dim1 lo
    g1[3] = (int)(((unsigned)(rows >> 16) & 0xFFFFu) |
                  (64u << 16));                            // dim1 hi | tile_dim0
    g1[4] = 64;                                            // tile_dim1=64, tile_dim2=0
    g1[5] = N;                                             // tensor_dim0_stride lo
    g1[6] = 0;
    g1[7] = 0;
    tdm_i32x4 z4;
    z4[0] = 0; z4[1] = 0; z4[2] = 0; z4[3] = 0;
    tdm_i32x8 z8;
    z8[0] = 0; z8[1] = 0; z8[2] = 0; z8[3] = 0;
    z8[4] = 0; z8[5] = 0; z8[6] = 0; z8[7] = 0;
    __builtin_amdgcn_tensor_load_to_lds(g0, g1, z4, z4, z8, 0);
}

// ---------------------------------------------------------------------------
// Generic WMMA GEMM:  C[M,N] = epi( A[M,K](f32) * WT[K,N](bf16) + bias[N] )
// WT is pre-transposed to K-major.  B panels are staged into LDS by the
// Tensor Data Mover (double-buffered 64K x 64N slabs); wave 0 issues the DMA
// for slab i+1 while all 8 waves run WMMAs on slab i (s_wait_tensorcnt +
// barrier pipelining).  A fragments are converted f32->bf16 on the fly.
// Each wave: 16 rows x 64 cols = 4 accum tiles.  Block = 8 waves = 128x64.
// grid = (N/64, M/128).  K must be a multiple of 64 (true for all call sites).
// ---------------------------------------------------------------------------
__global__ __launch_bounds__(256) void wmma_gemm_kernel(
    const float* __restrict__ A, int lda,
    const __bf16* __restrict__ WT, int N, int K,
    const float* __restrict__ bias,
    float* __restrict__ outF, int ldF,
    __bf16* __restrict__ outB, int ldB,
    float scale, int relu)
{
    __shared__ __bf16 smem[2][64 * 64];    // double-buffered B slab (2 x 8KB)

    const int wave = threadIdx.x >> 5;
    const int l    = threadIdx.x & 31;
    const int m16  = l & 15;
    const int g    = l >> 4;
    const int rowBase = blockIdx.y * 128 + wave * 16;
    const int nBase   = blockIdx.x * 64;

    const unsigned ldsBase = (unsigned)(uintptr_t)(&smem[0][0]);

    v8f acc[4];
#pragma unroll
    for (int t = 0; t < 4; ++t) acc[t] = (v8f)(0.0f);

    const float* arow = A + (size_t)(rowBase + m16) * lda;
    const int nSlab = K >> 6;

    // prologue: slab 0 -> buffer 0
    if (threadIdx.x < 32)
        tdm_load_slab(WT + nBase, ldsBase, N - nBase, K, N);

    for (int i = 0; i < nSlab; ++i) {
        if (threadIdx.x < 32) {
            if (i + 1 < nSlab) {
                tdm_load_slab(WT + (size_t)((i + 1) * 64) * N + nBase,
                              ldsBase + (unsigned)(((i + 1) & 1) * 64 * 64 * 2),
                              N - nBase, K - (i + 1) * 64, N);
                __builtin_amdgcn_s_wait_tensorcnt(1);   // slab i has landed
            } else {
                __builtin_amdgcn_s_wait_tensorcnt(0);
            }
        }
        __syncthreads();                                // slab i visible to WG

        const __bf16* sb = &smem[i & 1][0];
#pragma unroll
        for (int ks = 0; ks < 2; ++ks) {
            const int k0 = i * 64 + ks * 32;
            // ---- A fragment (f32 -> bf16), per-lane K pattern per ISA 7.12.2
            const float* ap = arow + k0 + g * 8;
            v4f x0 = *(const v4f*)(ap);
            v4f x1 = *(const v4f*)(ap + 4);
            v4f x2 = *(const v4f*)(ap + 16);
            v4f x3 = *(const v4f*)(ap + 20);
            v16bf a;
#pragma unroll
            for (int h = 0; h < 4; ++h) {
                a[h]      = (__bf16)x0[h];
                a[4 + h]  = (__bf16)x1[h];
                a[8 + h]  = (__bf16)x2[h];
                a[12 + h] = (__bf16)x3[h];
            }
            // ---- B fragments from LDS slab: lane owns N-col, 16 halves = K (table order)
            const __bf16* bp = sb + (size_t)(ks * 32 + g * 8) * 64 + m16;
#pragma unroll
            for (int t = 0; t < 4; ++t) {
                v16bf bfrag;
#pragma unroll
                for (int s = 0; s < 16; ++s)
                    bfrag[s] = bp[(size_t)((s & 7) + (s >> 3) * 16) * 64 + t * 16];
                acc[t] = __builtin_amdgcn_wmma_f32_16x16x32_bf16(
                    false, a, false, bfrag, (short)0, acc[t], false, false);
            }
        }
        __syncthreads();            // WG done with buf[i&1] before TDM reuses it
    }

    // ---- epilogue (C/D layout: M = r + 8*(l>>4), N = l&15)
#pragma unroll
    for (int t = 0; t < 4; ++t) {
#pragma unroll
        for (int r = 0; r < 8; ++r) {
            int mm = rowBase + r + 8 * g;
            int nn = nBase + t * 16 + m16;
            float v = acc[t][r] + bias[nn];
            v *= scale;
            if (relu) v = fmaxf(v, 0.0f);
            if (outF) outF[(size_t)mm * ldF + nn] = v;
            if (outB) outB[(size_t)mm * ldB + nn] = (__bf16)v;
        }
    }
}

// ---------------------------------------------------------------------------
// Single-head attention for one step: per-batch scores (q . K_s), softmax,
// ctx = sum_s a_s * V_s.  One workgroup per batch row; S in {256, 32}.
// K/V are bf16 (resident in L2 across all 256 steps).
// ---------------------------------------------------------------------------
__global__ __launch_bounds__(256) void attn_kernel(
    const float* __restrict__ q, const __bf16* __restrict__ Kb,
    const __bf16* __restrict__ Vb, float* __restrict__ ctx, int S)
{
    __shared__ float sc[256];
    __shared__ float red[256];
    const int b   = blockIdx.x;
    const int tid = threadIdx.x;
    const float* qr = q + (size_t)b * 512;

    float sv = -1e30f;
    if (tid < S) {
        const __bf16* kr = Kb + ((size_t)b * S + tid) * 512;
        float acc = 0.0f;
#pragma unroll 4
        for (int e = 0; e < 512; e += 8) {
            v8bf kv = *(const v8bf*)(kr + e);
            v4f q0 = *(const v4f*)(qr + e);
            v4f q1 = *(const v4f*)(qr + e + 4);
            acc += q0[0]*(float)kv[0] + q0[1]*(float)kv[1]
                 + q0[2]*(float)kv[2] + q0[3]*(float)kv[3]
                 + q1[0]*(float)kv[4] + q1[1]*(float)kv[5]
                 + q1[2]*(float)kv[6] + q1[3]*(float)kv[7];
        }
        sv = acc;
    }

    red[tid] = sv; __syncthreads();
    for (int off = 128; off > 0; off >>= 1) {
        if (tid < off) red[tid] = fmaxf(red[tid], red[tid + off]);
        __syncthreads();
    }
    float mx = red[0]; __syncthreads();

    float ev = (tid < S) ? __expf(sv - mx) : 0.0f;
    red[tid] = ev; __syncthreads();
    for (int off = 128; off > 0; off >>= 1) {
        if (tid < off) red[tid] += red[tid + off];
        __syncthreads();
    }
    float denom = red[0]; __syncthreads();
    sc[tid] = ev / denom;
    __syncthreads();

    for (int e = tid; e < 512; e += 256) {
        float acc = 0.0f;
        for (int s = 0; s < S; ++s)
            acc += sc[s] * (float)Vb[((size_t)b * S + s) * 512 + e];
        ctx[(size_t)b * 512 + e] = acc;
    }
}

// ---------------------------------------------------------------------------
// LSTM cell pointwise (torch gate order i,f,g,o) + stage next step's [x|h]
// ---------------------------------------------------------------------------
__global__ __launch_bounds__(256) void lstm_cell_kernel(
    const float* __restrict__ gates, float* __restrict__ c,
    float* __restrict__ xh, const float* __restrict__ tos, int t)
{
    int idx = blockIdx.x * 256 + threadIdx.x;     // 128*512 threads
    int b = idx >> 9, j = idx & 511;
    const float* gr = gates + (size_t)b * 2048;
    float ig = sigmoidf_(gr[j]);
    float fg = sigmoidf_(gr[512 + j]);
    float gg = tanhf(gr[1024 + j]);
    float og = sigmoidf_(gr[1536 + j]);
    float cn = fg * c[idx] + ig * gg;
    float h  = og * tanhf(cn);
    c[idx] = cn;
    xh[(size_t)b * 640 + 128 + j] = h;
    if (j < 128 && (t + 1) < 256)
        xh[(size_t)b * 640 + j] = tos[((size_t)b * 256 + (t + 1)) * 128 + j];
}

// h0/c0 = concat(hn[0], hn[1]); x0 = zeros
__global__ __launch_bounds__(256) void init_state_kernel(
    const float* __restrict__ hn, const float* __restrict__ cn,
    float* __restrict__ c, float* __restrict__ xh)
{
    int idx = blockIdx.x * 256 + threadIdx.x;     // 128*512 threads
    int b = idx >> 9, j = idx & 511;
    int src = (j >= 256 ? 128 * 256 : 0) + b * 256 + (j & 255);
    c[idx] = cn[src];
    xh[(size_t)b * 640 + 128 + j] = hn[src];
    if (j < 128) xh[(size_t)b * 640 + j] = 0.0f;
}

// W[N,K] f32  ->  WT[K,N] bf16
__global__ __launch_bounds__(256) void transpose_bf16_kernel(
    const float* __restrict__ W, __bf16* __restrict__ WT, int N, int K)
{
    int idx = blockIdx.x * 256 + threadIdx.x;
    if (idx >= N * K) return;
    int n = idx % N, k = idx / N;
    WT[idx] = (__bf16)W[(size_t)n * K + k];
}

// [Wih | Whh] transposed:  WT[k, n],  k<128 -> Wih, else Whh
__global__ __launch_bounds__(256) void build_wcat_kernel(
    const float* __restrict__ Wih, const float* __restrict__ Whh,
    __bf16* __restrict__ WT)
{
    int idx = blockIdx.x * 256 + threadIdx.x;     // 640*2048
    int n = idx & 2047, k = idx >> 11;
    float v = (k < 128) ? Wih[(size_t)n * 128 + k]
                        : Whh[(size_t)n * 512 + (k - 128)];
    WT[idx] = (__bf16)v;
}

__global__ __launch_bounds__(256) void bias_sum_kernel(
    const float* __restrict__ a, const float* __restrict__ b,
    float* __restrict__ o, int n)
{
    int i = blockIdx.x * 256 + threadIdx.x;
    if (i < n) o[i] = a[i] + b[i];
}

// ---------------------------------------------------------------------------
// Host-side orchestration
// ---------------------------------------------------------------------------
static inline void gemm(hipStream_t s, const float* A, int lda,
                        const __bf16* WT, int N, int K, const float* bias,
                        float* outF, int ldF, __bf16* outB, int ldB,
                        int M, float scale, int relu)
{
    dim3 grid(N / 64, M / 128);
    wmma_gemm_kernel<<<grid, 256, 0, s>>>(A, lda, WT, N, K, bias,
                                          outF, ldF, outB, ldB, scale, relu);
}

extern "C" void kernel_launch(void* const* d_in, const int* in_sizes, int n_in,
                              void* d_out, int out_size, void* d_ws, size_t ws_size,
                              hipStream_t stream)
{
    (void)in_sizes; (void)n_in; (void)out_size; (void)ws_size;

    const float* char_enc = (const float*)d_in[0];   // [128,256,512]
    const float* char_hn  = (const float*)d_in[1];   // [2,128,256]
    const float* char_cn  = (const float*)d_in[2];   // [2,128,256]
    const float* tag_enc  = (const float*)d_in[3];   // [128,32,512]
    const float* tos      = (const float*)d_in[4];   // [128,256,128]
    const float* lstm_Wih = (const float*)d_in[5];   // [2048,128]
    const float* lstm_Whh = (const float*)d_in[6];   // [2048,512]
    const float* lstm_bih = (const float*)d_in[7];
    const float* lstm_bhh = (const float*)d_in[8];
    const float* ca_Wq = (const float*)d_in[9];
    const float* ca_Wk = (const float*)d_in[10];
    const float* ca_Wv = (const float*)d_in[11];
    const float* ca_bq = (const float*)d_in[12];
    const float* ca_bk = (const float*)d_in[13];
    const float* ca_bv = (const float*)d_in[14];
    const float* ca_Wo = (const float*)d_in[15];
    const float* ca_bo = (const float*)d_in[16];
    const float* ta_Wq = (const float*)d_in[17];
    const float* ta_Wk = (const float*)d_in[18];
    const float* ta_Wv = (const float*)d_in[19];
    const float* ta_bq = (const float*)d_in[20];
    const float* ta_bk = (const float*)d_in[21];
    const float* ta_bv = (const float*)d_in[22];
    const float* ta_Wo = (const float*)d_in[23];
    const float* ta_bo = (const float*)d_in[24];
    const float* out_W = (const float*)d_in[25];     // [128,1024]
    const float* out_b = (const float*)d_in[26];
    float* out = (float*)d_out;                      // [128,256,128]

    // workspace carve-up
    char*  ws  = (char*)d_ws;
    size_t off = 0;
    auto alloc = [&](size_t bytes) -> void* {
        void* p = ws + off;
        off += (bytes + 255) & ~(size_t)255;
        return p;
    };
    __bf16* KcB    = (__bf16*)alloc((size_t)128 * 256 * 512 * 2);
    __bf16* VcB    = (__bf16*)alloc((size_t)128 * 256 * 512 * 2);
    __bf16* KtB    = (__bf16*)alloc((size_t)128 * 32 * 512 * 2);
    __bf16* VtB    = (__bf16*)alloc((size_t)128 * 32 * 512 * 2);
    __bf16* WcatT  = (__bf16*)alloc((size_t)640 * 2048 * 2);
    __bf16* caWqT  = (__bf16*)alloc((size_t)512 * 512 * 2);
    __bf16* caWkT  = (__bf16*)alloc((size_t)512 * 512 * 2);
    __bf16* caWvT  = (__bf16*)alloc((size_t)512 * 512 * 2);
    __bf16* caWoT  = (__bf16*)alloc((size_t)512 * 512 * 2);
    __bf16* taWqT  = (__bf16*)alloc((size_t)512 * 512 * 2);
    __bf16* taWkT  = (__bf16*)alloc((size_t)512 * 512 * 2);
    __bf16* taWvT  = (__bf16*)alloc((size_t)512 * 512 * 2);
    __bf16* taWoT  = (__bf16*)alloc((size_t)512 * 512 * 2);
    __bf16* outWT  = (__bf16*)alloc((size_t)1024 * 128 * 2);
    float*  bsum   = (float*)alloc(2048 * 4);
    float*  xh     = (float*)alloc((size_t)128 * 640 * 4);   // [x_t | h]
    float*  cbuf   = (float*)alloc((size_t)128 * 512 * 4);
    float*  gates  = (float*)alloc((size_t)128 * 2048 * 4);
    float*  qbuf   = (float*)alloc((size_t)128 * 512 * 4);
    float*  ctxbuf = (float*)alloc((size_t)128 * 512 * 4);
    float*  agg    = (float*)alloc((size_t)128 * 1024 * 4);

    const int TB = 256;
    auto blocks = [](int n) { return (n + 255) / 256; };

    // ---- weight prep (transpose + f32->bf16) -----------------------------
    transpose_bf16_kernel<<<blocks(512*512), TB, 0, stream>>>(ca_Wq, caWqT, 512, 512);
    transpose_bf16_kernel<<<blocks(512*512), TB, 0, stream>>>(ca_Wk, caWkT, 512, 512);
    transpose_bf16_kernel<<<blocks(512*512), TB, 0, stream>>>(ca_Wv, caWvT, 512, 512);
    transpose_bf16_kernel<<<blocks(512*512), TB, 0, stream>>>(ca_Wo, caWoT, 512, 512);
    transpose_bf16_kernel<<<blocks(512*512), TB, 0, stream>>>(ta_Wq, taWqT, 512, 512);
    transpose_bf16_kernel<<<blocks(512*512), TB, 0, stream>>>(ta_Wk, taWkT, 512, 512);
    transpose_bf16_kernel<<<blocks(512*512), TB, 0, stream>>>(ta_Wv, taWvT, 512, 512);
    transpose_bf16_kernel<<<blocks(512*512), TB, 0, stream>>>(ta_Wo, taWoT, 512, 512);
    transpose_bf16_kernel<<<blocks(128*1024), TB, 0, stream>>>(out_W, outWT, 128, 1024);
    build_wcat_kernel<<<blocks(640*2048), TB, 0, stream>>>(lstm_Wih, lstm_Whh, WcatT);
    bias_sum_kernel<<<blocks(2048), TB, 0, stream>>>(lstm_bih, lstm_bhh, bsum, 2048);

    // ---- hoisted K/V projections (WMMA, bf16 outputs live in L2) ---------
    gemm(stream, char_enc, 512, caWkT, 512, 512, ca_bk, nullptr, 0, KcB, 512, 32768, 1.0f, 0);
    gemm(stream, char_enc, 512, caWvT, 512, 512, ca_bv, nullptr, 0, VcB, 512, 32768, 1.0f, 0);
    gemm(stream, tag_enc,  512, taWkT, 512, 512, ta_bk, nullptr, 0, KtB, 512, 4096,  1.0f, 0);
    gemm(stream, tag_enc,  512, taWvT, 512, 512, ta_bv, nullptr, 0, VtB, 512, 4096,  1.0f, 0);

    // ---- initial state ---------------------------------------------------
    init_state_kernel<<<256, TB, 0, stream>>>(char_hn, char_cn, cbuf, xh);

    const float QSCALE = 0.044194173824159216f;  // 1/sqrt(512)

    // ---- 256 sequential decode steps ------------------------------------
    for (int t = 0; t < 256; ++t) {
        // gates = [x_t|h] @ [Wih|Whh]^T + (bih+bhh)
        gemm(stream, xh, 640, WcatT, 2048, 640, bsum, gates, 2048, nullptr, 0, 128, 1.0f, 0);
        lstm_cell_kernel<<<256, TB, 0, stream>>>(gates, cbuf, xh, tos, t);

        // char attention
        gemm(stream, xh + 128, 640, caWqT, 512, 512, ca_bq, qbuf, 512, nullptr, 0, 128, QSCALE, 0);
        attn_kernel<<<128, TB, 0, stream>>>(qbuf, KcB, VcB, ctxbuf, 256);
        gemm(stream, ctxbuf, 512, caWoT, 512, 512, ca_bo, agg, 1024, nullptr, 0, 128, 1.0f, 1);

        // tag attention
        gemm(stream, xh + 128, 640, taWqT, 512, 512, ta_bq, qbuf, 512, nullptr, 0, 128, QSCALE, 0);
        attn_kernel<<<128, TB, 0, stream>>>(qbuf, KtB, VtB, ctxbuf, 32);
        gemm(stream, ctxbuf, 512, taWoT, 512, 512, ta_bo, agg + 512, 1024, nullptr, 0, 128, 1.0f, 1);

        // logits for step t -> d_out[:, t, :]
        gemm(stream, agg, 1024, outWT, 128, 1024, out_b,
             out + (size_t)t * 128, 256 * 128, nullptr, 0, 128, 1.0f, 0);
    }
}
